// GRAPELayer_27144193311129
// MI455X (gfx1250) — matrix operand order repaired and checked by reading, not compiled, not run on hardware
//
#include <hip/hip_runtime.h>
#include <hip/hip_bf16.h>

// ---------------------------------------------------------------------------
// GRAPE GNN layer for MI455X (gfx1250): bf16 WMMA GEMMs, LDS-staged weights,
// L2-resident node-feature gathers, f32 atomic scatter-mean.
// Round 2: native v_cvt bf16 conversion; uniform full-tile fast paths to kill
// per-element EXEC-mask predication in the steady state.
// ---------------------------------------------------------------------------

typedef __attribute__((ext_vector_type(16))) __bf16 v16bf;
typedef __attribute__((ext_vector_type(8)))  float  v8f;

#define NDIM 128   // NODE_DIM == EDGE_DIM == 128

union AFrag {
    v16bf v;
    __bf16 s[16];
    uint4 q[2];
};

// Load 8 contiguous f32 and convert to bf16 (native v_cvt_pk_bf16_f32 path)
__device__ __forceinline__ void load8_bf16(__bf16* d, const float* s) {
    float4 a = *reinterpret_cast<const float4*>(s);
    float4 b = *reinterpret_cast<const float4*>(s + 4);
    d[0] = (__bf16)a.x; d[1] = (__bf16)a.y; d[2] = (__bf16)a.z; d[3] = (__bf16)a.w;
    d[4] = (__bf16)b.x; d[5] = (__bf16)b.y; d[6] = (__bf16)b.z; d[7] = (__bf16)b.w;
}

__device__ __forceinline__ void load8_bf16_scaled(__bf16* d, const float* s, float r) {
    float4 a = *reinterpret_cast<const float4*>(s);
    float4 b = *reinterpret_cast<const float4*>(s + 4);
    d[0] = (__bf16)(a.x * r); d[1] = (__bf16)(a.y * r);
    d[2] = (__bf16)(a.z * r); d[3] = (__bf16)(a.w * r);
    d[4] = (__bf16)(b.x * r); d[5] = (__bf16)(b.y * r);
    d[6] = (__bf16)(b.z * r); d[7] = (__bf16)(b.w * r);
}

// B fragment: 512 bf16 per (ntile, kchunk); lane holds column N = nt*16 + (lane&15),
// elements j -> K = kc*32 + 16*(j>>3) + 8*(lane>>4) + (j&7).
__device__ __forceinline__ v16bf load_bfrag(const unsigned short* base, int frag, int lane) {
    union { v16bf v; uint4 q[2]; } u;
    const uint4* p = reinterpret_cast<const uint4*>(base + frag * 512 + lane * 16);
    u.q[0] = p[0];
    u.q[1] = p[1];
    return u.v;
}

__device__ __forceinline__ unsigned short f32_to_bf16_bits(float f) {
    __bf16 b = (__bf16)f;
    return __builtin_bit_cast(unsigned short, b);
}

// ---------------------------------------------------------------------------
// Kernel 1: pack f32 weights [K x 128] row-major into bf16 WMMA B fragments.
// Fragment index = nt * (K/32) + kc ; within: lane*16 + j.
// ---------------------------------------------------------------------------
__global__ void grape_pack_weights(const float* __restrict__ Pw,
                                   const float* __restrict__ Qw,
                                   const float* __restrict__ Ww,
                                   unsigned short* __restrict__ pP,
                                   unsigned short* __restrict__ pQ,
                                   unsigned short* __restrict__ pW) {
    const int total = (256 + 256 + 384) * 128;
    for (int idx = blockIdx.x * blockDim.x + threadIdx.x; idx < total;
         idx += gridDim.x * blockDim.x) {
        const float* src; unsigned short* dst; int K; int local;
        if (idx < 256 * 128)          { src = Pw; dst = pP; K = 256; local = idx; }
        else if (idx < 2 * 256 * 128) { src = Qw; dst = pQ; K = 256; local = idx - 256 * 128; }
        else                          { src = Ww; dst = pW; K = 384; local = idx - 2 * 256 * 128; }
        int nk   = K >> 5;
        int frag = local >> 9;
        int wi   = local & 511;
        int lane = wi >> 4;
        int j    = wi & 15;
        int nt   = frag / nk;
        int kc   = frag % nk;
        int k    = kc * 32 + ((j >> 3) << 4) + ((lane >> 4) << 3) + (j & 7);
        int n    = nt * 16 + (lane & 15);
        dst[local] = f32_to_bf16_bits(src[k * 128 + n]);
    }
}

// ---------------------------------------------------------------------------
// Kernel 2: per-edge. One wave computes a 16-edge x 128-col tile.
//   messages = relu([h_u | e] @ P + Pb)  -> atomic scatter to agg[tgt], deg++
//   e_new    = relu([e | h_u | h_v] @ W + Wb)
// Weights staged in LDS (P: 64KB, W: 96KB => 160KB of the 320KB WGP LDS).
// ---------------------------------------------------------------------------
__global__ __launch_bounds__(256) void grape_edge_kernel(
    const float* __restrict__ h, const float* __restrict__ e,
    const int* __restrict__ ei,                       // [2*E]: src then tgt
    const float* __restrict__ Pb, const float* __restrict__ Wb,
    const unsigned short* __restrict__ pP, const unsigned short* __restrict__ pW,
    float* __restrict__ agg, float* __restrict__ deg,
    float* __restrict__ e_out, int E, int nTiles) {

    extern __shared__ unsigned short lds[];
    unsigned short* sP = lds;               // 256*128 = 32768 ushorts
    unsigned short* sW = lds + 32768;       // 384*128 = 49152 ushorts
    {
        const uint4* gp = reinterpret_cast<const uint4*>(pP);
        uint4*       sp = reinterpret_cast<uint4*>(sP);
        for (int i = threadIdx.x; i < 32768 / 8; i += blockDim.x) sp[i] = gp[i];
        const uint4* gw = reinterpret_cast<const uint4*>(pW);
        uint4*       sw = reinterpret_cast<uint4*>(sW);
        for (int i = threadIdx.x; i < 49152 / 8; i += blockDim.x) sw[i] = gw[i];
    }
    __syncthreads();

    const int lane = threadIdx.x & 31;
    const int wave = threadIdx.x >> 5;
    const int row  = lane & 15;     // A-matrix row / C-matrix column N
    const int half = lane >> 4;

    for (int tile = blockIdx.x * 8 + wave; tile < nTiles; tile += gridDim.x * 8) {
        const bool fullTile = (tile * 16 + 16) <= E;   // uniform across the wave

        int edge = tile * 16 + row;
        if (edge >= E) edge = E - 1;                   // clamp loads, predicate stores
        const int src = ei[edge];
        const int tgt = ei[E + edge];
        const float* hu = h + (long long)src  * NDIM;
        const float* hv = h + (long long)tgt  * NDIM;
        const float* er = e + (long long)edge * NDIM;

        // target row indices for the scatter (element i lives at row M = i + 8*half)
        int trow[8];
#pragma unroll
        for (int i = 0; i < 8; ++i) {
            int em = tile * 16 + i + 8 * half;
            int emc = em < E ? em : E - 1;
            trow[i] = ei[E + emc];
        }

        // ----- message GEMM: A = [h_u | e], K = 256 -----
        v8f acc[8];
#pragma unroll
        for (int nt = 0; nt < 8; ++nt) { v8f z = {}; acc[nt] = z; }
        for (int kc = 0; kc < 8; ++kc) {
            const float* arow = (kc < 4) ? hu : er;
            int c0 = ((kc < 4) ? kc : kc - 4) * 32 + half * 8;
            AFrag af;
            load8_bf16(af.s,     arow + c0);
            load8_bf16(af.s + 8, arow + c0 + 16);
#pragma unroll
            for (int nt = 0; nt < 8; ++nt) {
                v16bf bf = load_bfrag(sP, nt * 8 + kc, lane);
                acc[nt] = __builtin_amdgcn_wmma_f32_16x16x32_bf16(
                    false, af.v, false, bf, (short)0, acc[nt], false, false);
            }
        }
        // bias + relu + atomic scatter of messages
        if (fullTile) {
#pragma unroll
            for (int nt = 0; nt < 8; ++nt) {
                float b = Pb[nt * 16 + row];
#pragma unroll
                for (int i = 0; i < 8; ++i) {
                    float v = fmaxf(acc[nt][i] + b, 0.0f);
                    atomicAdd(&agg[(long long)trow[i] * NDIM + nt * 16 + row], v);
                }
            }
            if (half == 0) atomicAdd(&deg[tgt], 1.0f);
        } else {
#pragma unroll
            for (int nt = 0; nt < 8; ++nt) {
                float b = Pb[nt * 16 + row];
#pragma unroll
                for (int i = 0; i < 8; ++i) {
                    float v = fmaxf(acc[nt][i] + b, 0.0f);
                    if (tile * 16 + i + 8 * half < E)
                        atomicAdd(&agg[(long long)trow[i] * NDIM + nt * 16 + row], v);
                }
            }
            if (half == 0 && (tile * 16 + row) < E) atomicAdd(&deg[tgt], 1.0f);
        }

        // ----- edge GEMM: A = [e | h_u | h_v], K = 384 -----
#pragma unroll
        for (int nt = 0; nt < 8; ++nt) { v8f z = {}; acc[nt] = z; }
        for (int kc = 0; kc < 12; ++kc) {
            const float* arow; int c0;
            if (kc < 4)      { arow = er; c0 = kc * 32; }
            else if (kc < 8) { arow = hu; c0 = (kc - 4) * 32; }
            else             { arow = hv; c0 = (kc - 8) * 32; }
            c0 += half * 8;
            AFrag af;
            load8_bf16(af.s,     arow + c0);
            load8_bf16(af.s + 8, arow + c0 + 16);
#pragma unroll
            for (int nt = 0; nt < 8; ++nt) {
                v16bf bf = load_bfrag(sW, nt * 12 + kc, lane);
                acc[nt] = __builtin_amdgcn_wmma_f32_16x16x32_bf16(
                    false, af.v, false, bf, (short)0, acc[nt], false, false);
            }
        }
        if (fullTile) {
#pragma unroll
            for (int nt = 0; nt < 8; ++nt) {
                float b = Wb[nt * 16 + row];
#pragma unroll
                for (int i = 0; i < 8; ++i) {
                    int em = tile * 16 + i + 8 * half;
                    float v = fmaxf(acc[nt][i] + b, 0.0f);
                    e_out[(long long)em * NDIM + nt * 16 + row] = v;
                }
            }
        } else {
#pragma unroll
            for (int nt = 0; nt < 8; ++nt) {
                float b = Wb[nt * 16 + row];
#pragma unroll
                for (int i = 0; i < 8; ++i) {
                    int em = tile * 16 + i + 8 * half;
                    float v = fmaxf(acc[nt][i] + b, 0.0f);
                    if (em < E)
                        e_out[(long long)em * NDIM + nt * 16 + row] = v;
                }
            }
        }
    }
}

// ---------------------------------------------------------------------------
// Kernel 3: per-node.  h_new = relu([h | agg/max(deg,1)] @ Q + Qb)
// ---------------------------------------------------------------------------
__global__ __launch_bounds__(256) void grape_node_kernel(
    const float* __restrict__ h, const float* __restrict__ agg,
    const float* __restrict__ deg, const float* __restrict__ Qb,
    const unsigned short* __restrict__ pQ,
    float* __restrict__ h_out, int N, int nTiles) {

    extern __shared__ unsigned short sQ[];  // 256*128 = 32768 ushorts = 64KB
    {
        const uint4* gq = reinterpret_cast<const uint4*>(pQ);
        uint4*       sq = reinterpret_cast<uint4*>(sQ);
        for (int i = threadIdx.x; i < 32768 / 8; i += blockDim.x) sq[i] = gq[i];
    }
    __syncthreads();

    const int lane = threadIdx.x & 31;
    const int wave = threadIdx.x >> 5;
    const int row  = lane & 15;
    const int half = lane >> 4;

    for (int tile = blockIdx.x * 8 + wave; tile < nTiles; tile += gridDim.x * 8) {
        const bool fullTile = (tile * 16 + 16) <= N;

        int node = tile * 16 + row;
        if (node >= N) node = N - 1;
        const float rdeg = 1.0f / fmaxf(deg[node], 1.0f);
        const float* hr = h   + (long long)node * NDIM;
        const float* ar = agg + (long long)node * NDIM;

        v8f acc[8];
#pragma unroll
        for (int nt = 0; nt < 8; ++nt) { v8f z = {}; acc[nt] = z; }
        for (int kc = 0; kc < 8; ++kc) {
            AFrag af;
            if (kc < 4) {
                int c0 = kc * 32 + half * 8;
                load8_bf16(af.s,     hr + c0);
                load8_bf16(af.s + 8, hr + c0 + 16);
            } else {
                int c0 = (kc - 4) * 32 + half * 8;
                load8_bf16_scaled(af.s,     ar + c0,      rdeg);
                load8_bf16_scaled(af.s + 8, ar + c0 + 16, rdeg);
            }
#pragma unroll
            for (int nt = 0; nt < 8; ++nt) {
                v16bf bf = load_bfrag(sQ, nt * 8 + kc, lane);
                acc[nt] = __builtin_amdgcn_wmma_f32_16x16x32_bf16(
                    false, af.v, false, bf, (short)0, acc[nt], false, false);
            }
        }
        if (fullTile) {
#pragma unroll
            for (int nt = 0; nt < 8; ++nt) {
                float b = Qb[nt * 16 + row];
#pragma unroll
                for (int i = 0; i < 8; ++i) {
                    int nm = tile * 16 + i + 8 * half;
                    float v = fmaxf(acc[nt][i] + b, 0.0f);
                    h_out[(long long)nm * NDIM + nt * 16 + row] = v;
                }
            }
        } else {
#pragma unroll
            for (int nt = 0; nt < 8; ++nt) {
                float b = Qb[nt * 16 + row];
#pragma unroll
                for (int i = 0; i < 8; ++i) {
                    int nm = tile * 16 + i + 8 * half;
                    float v = fmaxf(acc[nt][i] + b, 0.0f);
                    if (nm < N)
                        h_out[(long long)nm * NDIM + nt * 16 + row] = v;
                }
            }
        }
    }
}

// ---------------------------------------------------------------------------
extern "C" void kernel_launch(void* const* d_in, const int* in_sizes, int n_in,
                              void* d_out, int out_size, void* d_ws, size_t ws_size,
                              hipStream_t stream) {
    const float* h  = (const float*)d_in[0];
    const float* e  = (const float*)d_in[1];
    const int*   ei = (const int*)d_in[2];
    const float* Pw = (const float*)d_in[3];
    const float* Pb = (const float*)d_in[4];
    const float* Qw = (const float*)d_in[5];
    const float* Qb = (const float*)d_in[6];
    const float* Ww = (const float*)d_in[7];
    const float* Wb = (const float*)d_in[8];

    const int N = in_sizes[0] / NDIM;       // 100000
    const int E = in_sizes[1] / NDIM;       // 625000

    // workspace layout: [agg f32 N*128][deg f32 N][pP bf16 32768][pQ 32768][pW 49152]
    char* ws = (char*)d_ws;
    float* agg = (float*)ws;
    float* deg = (float*)(ws + (size_t)N * NDIM * sizeof(float));
    unsigned short* pP = (unsigned short*)(ws + (size_t)N * NDIM * sizeof(float)
                                              + (size_t)N * sizeof(float));
    unsigned short* pQ = pP + 256 * 128;
    unsigned short* pW = pQ + 256 * 128;

    float* h_out = (float*)d_out;
    float* e_out = h_out + (size_t)N * NDIM;

    // zero the scatter accumulators + degrees (ws is poisoned by the harness)
    hipMemsetAsync(agg, 0, (size_t)N * NDIM * sizeof(float) + (size_t)N * sizeof(float),
                   stream);

    grape_pack_weights<<<112, 256, 0, stream>>>(Pw, Qw, Ww, pP, pQ, pW);

    const int nTilesE = (E + 15) / 16;
    const int nTilesN = (N + 15) / 16;
    grape_edge_kernel<<<1280, 256, (32768 + 49152) * sizeof(unsigned short), stream>>>(
        h, e, ei, Pb, Wb, pP, pW, agg, deg, e_out, E, nTilesE);
    grape_node_kernel<<<800, 256, 32768 * sizeof(unsigned short), stream>>>(
        h, agg, deg, Qb, pQ, h_out, N, nTilesN);
}
